// GNNV2Anomaly_18674517803533
// MI455X (gfx1250) — compile-verified, matrix operand order checked by reading, not run on hardware
//
#include <hip/hip_runtime.h>
#include <hip/hip_bf16.h>
#include <math.h>

#define NN 100000
#define EE 1600000
#define DD 128
#define LL 4
#define ETOT (EE + NN)
#define ROWS 128   // rows per GEMM block (8 M-tiles per wave)

typedef __attribute__((ext_vector_type(16))) __bf16 v16bf;
typedef __attribute__((ext_vector_type(8)))  float  v8f;
typedef __attribute__((ext_vector_type(4)))  float  v4f;
typedef __attribute__((ext_vector_type(4)))  unsigned v4u;

union Frag16 { v16bf v; v4u q[2]; };

__device__ __forceinline__ unsigned short f2bf(float f) {
  unsigned u = __float_as_uint(f);
  unsigned r = u + 0x7FFFu + ((u >> 16) & 1u);   // round-to-nearest-even
  return (unsigned short)(r >> 16);
}

__device__ __forceinline__ unsigned f2ord(float f) {
  unsigned u = __float_as_uint(f);
  return (u & 0x80000000u) ? ~u : (u | 0x80000000u);
}
__device__ __forceinline__ float ord2f(unsigned u) {
  return __uint_as_float((u & 0x80000000u) ? (u & 0x7FFFFFFFu) : ~u);
}

// C[nrows x 128] = A[nrows x 128] @ W[128 x 128] (+ bias), bf16 WMMA, fp32 acc.
// One block = 128 rows; 8 waves, each wave owns one 16-column tile across 8 M-tiles.
__global__ __launch_bounds__(256) void gemm128(const float* __restrict__ A,
                                               const float* __restrict__ W,
                                               const float* __restrict__ bias,
                                               float* __restrict__ C, int nrows) {
  __shared__ unsigned short Ald[ROWS * 128];  // 32 KB: A stripe, bf16
  __shared__ unsigned short Wtl[128 * 128];   // 32 KB: W transposed, bf16
  const int tid = threadIdx.x;
  const int rowBase = blockIdx.x * ROWS;
  const bool fullBlock = (rowBase + ROWS) <= nrows;

  // Stage A tile (f32 -> bf16)
  if (fullBlock) {
    for (int i = tid; i < ROWS * 32; i += 256) {
      int r = i >> 5, c4 = (i & 31) << 2;
      v4f v = *(const v4f*)(A + (size_t)(rowBase + r) * 128 + c4);
      *(unsigned*)&Ald[r * 128 + c4]     = ((unsigned)f2bf(v.y) << 16) | f2bf(v.x);
      *(unsigned*)&Ald[r * 128 + c4 + 2] = ((unsigned)f2bf(v.w) << 16) | f2bf(v.z);
    }
  } else {
    for (int i = tid; i < ROWS * 32; i += 256) {
      int r = i >> 5, c4 = (i & 31) << 2;
      v4f v = {0.f, 0.f, 0.f, 0.f};
      if (rowBase + r < nrows) v = *(const v4f*)(A + (size_t)(rowBase + r) * 128 + c4);
      *(unsigned*)&Ald[r * 128 + c4]     = ((unsigned)f2bf(v.y) << 16) | f2bf(v.x);
      *(unsigned*)&Ald[r * 128 + c4 + 2] = ((unsigned)f2bf(v.w) << 16) | f2bf(v.z);
    }
  }
  // Stage W transposed (so B fragment loads are contiguous b128 per lane)
  for (int i = tid; i < 128 * 32; i += 256) {
    int k = i >> 5, c4 = (i & 31) << 2;
    v4f v = *(const v4f*)(W + (size_t)k * 128 + c4);
    Wtl[(c4 + 0) * 128 + k] = f2bf(v.x);
    Wtl[(c4 + 1) * 128 + k] = f2bf(v.y);
    Wtl[(c4 + 2) * 128 + k] = f2bf(v.z);
    Wtl[(c4 + 3) * 128 + k] = f2bf(v.w);
  }
  __syncthreads();

  const int wave = tid >> 5, lane = tid & 31;
  const int half = lane >> 4, lm = lane & 15;
  const int colBase = wave * 16;

  // B fragments for all 4 K-steps, register-resident, shared across 8 M-tiles
  Frag16 Bf[4];
#pragma unroll
  for (int ks = 0; ks < 4; ++ks) {
    const unsigned short* p = &Wtl[(colBase + lm) * 128 + ks * 32 + half * 8];
    Bf[ks].q[0] = *(const v4u*)p;
    Bf[ks].q[1] = *(const v4u*)(p + 16);
  }
  const float bval = bias ? bias[colBase + lm] : 0.f;
  const int col = colBase + lm;

#pragma unroll
  for (int mt = 0; mt < ROWS / 16; ++mt) {
    v8f acc = {0.f, 0.f, 0.f, 0.f, 0.f, 0.f, 0.f, 0.f};
#pragma unroll
    for (int ks = 0; ks < 4; ++ks) {
      Frag16 Af;
      const unsigned short* p = &Ald[(mt * 16 + lm) * 128 + ks * 32 + half * 8];
      Af.q[0] = *(const v4u*)p;
      Af.q[1] = *(const v4u*)(p + 16);
      acc = __builtin_amdgcn_wmma_f32_16x16x32_bf16(false, Af.v, false, Bf[ks].v,
                                                    (short)0, acc, false, false);
    }
    const int rtile = rowBase + mt * 16 + half * 8;
    float* Cp = C + (size_t)rtile * 128 + col;
    if (fullBlock) {
      // Fast path: single base pointer, immediate row offsets, no per-element guards.
#pragma unroll
      for (int j = 0; j < 8; ++j) Cp[j * 128] = acc[j] + bval;
    } else {
#pragma unroll
      for (int j = 0; j < 8; ++j)
        if (rtile + j < nrows) Cp[j * 128] = acc[j] + bval;
    }
  }
}

// Seed h_next with per-column bias (makes aggregation bias free); zero max/denom.
__global__ __launch_bounds__(256) void layer_init(const float* __restrict__ b,
                                                  float* __restrict__ hn,
                                                  unsigned* __restrict__ mord,
                                                  float* __restrict__ denom) {
  int i = blockIdx.x * 256 + threadIdx.x;
  if (i < NN * DD) hn[i] = b[i & 127];
  if (i < NN) { mord[i] = 0u; denom[i] = 0.f; }
}

// One wave per edge: e = att . leakyrelu(xl[src]+xr[dst]); segment max via ordered atomicMax.
__global__ __launch_bounds__(256) void edge_scores(const float* __restrict__ xl,
                                                   const float* __restrict__ xr,
                                                   const float* __restrict__ att,
                                                   const int* __restrict__ src,
                                                   const int* __restrict__ dst,
                                                   float* __restrict__ ew,
                                                   unsigned* __restrict__ mord) {
  int wid  = (blockIdx.x * 256 + threadIdx.x) >> 5;
  int lane = threadIdx.x & 31;
  if (wid >= ETOT) return;
  int s, d;
  if (wid < EE) { s = src[wid]; d = dst[wid]; } else { s = d = wid - EE; }
  v4f a = *(const v4f*)(xl + (size_t)s * 128 + lane * 4);
  v4f r = *(const v4f*)(xr + (size_t)d * 128 + lane * 4);
  v4f t = *(const v4f*)(att + lane * 4);
  float v, sum = 0.f;
  v = a.x + r.x; v = v > 0.f ? v : 0.2f * v; sum += t.x * v;
  v = a.y + r.y; v = v > 0.f ? v : 0.2f * v; sum += t.y * v;
  v = a.z + r.z; v = v > 0.f ? v : 0.2f * v; sum += t.z * v;
  v = a.w + r.w; v = v > 0.f ? v : 0.2f * v; sum += t.w * v;
#pragma unroll
  for (int off = 16; off > 0; off >>= 1) sum += __shfl_xor(sum, off, 32);
  if (lane == 0) { ew[wid] = sum; atomicMax(&mord[d], f2ord(sum)); }
}

__global__ __launch_bounds__(256) void edge_norm(const int* __restrict__ dst,
                                                 const unsigned* __restrict__ mord,
                                                 float* __restrict__ ew,
                                                 float* __restrict__ denom) {
  int i = blockIdx.x * 256 + threadIdx.x;
  if (i >= ETOT) return;
  int d = (i < EE) ? dst[i] : (i - EE);
  float w = expf(ew[i] - ord2f(mord[d]));
  ew[i] = w;
  atomicAdd(&denom[d], w);
}

// One wave per edge: h_next[dst] += alpha * xl[src] (4 float atomics per lane, L2-resident).
__global__ __launch_bounds__(256) void edge_agg(const float* __restrict__ xl,
                                                const int* __restrict__ src,
                                                const int* __restrict__ dst,
                                                const float* __restrict__ ew,
                                                const float* __restrict__ denom,
                                                float* __restrict__ hn) {
  int wid  = (blockIdx.x * 256 + threadIdx.x) >> 5;
  int lane = threadIdx.x & 31;
  if (wid >= ETOT) return;
  int s, d;
  if (wid < EE) { s = src[wid]; d = dst[wid]; } else { s = d = wid - EE; }
  float alpha = ew[wid] / denom[d];
  v4f a = *(const v4f*)(xl + (size_t)s * 128 + lane * 4);
  float* o = hn + (size_t)d * 128 + lane * 4;
  atomicAdd(o + 0, alpha * a.x);
  atomicAdd(o + 1, alpha * a.y);
  atomicAdd(o + 2, alpha * a.z);
  atomicAdd(o + 3, alpha * a.w);
}

__global__ __launch_bounds__(256) void gelu_k(const float* __restrict__ in,
                                              float* __restrict__ out, int n) {
  int i = blockIdx.x * 256 + threadIdx.x;
  if (i >= n) return;
  float x = in[i];
  out[i] = 0.5f * x * (1.f + erff(x * 0.70710678118654752f));   // exact erf gelu
}

__global__ __launch_bounds__(256) void vae_k(const float* __restrict__ mu,
                                             float* __restrict__ ls,
                                             const float* __restrict__ eps,
                                             float* __restrict__ z, int n) {
  int i = blockIdx.x * 256 + threadIdx.x;
  if (i >= n) return;
  float l = fminf(ls[i], 10.f);
  ls[i] = l;
  z[i] = mu[i] + eps[i] * expf(l);
}

// degree_pred = z @ W_deg + b_deg : one wave per row.
__global__ __launch_bounds__(256) void deg_k(const float* __restrict__ z,
                                             const float* __restrict__ Wd,
                                             const float* __restrict__ bd,
                                             float* __restrict__ out) {
  int wid  = (blockIdx.x * 256 + threadIdx.x) >> 5;
  int lane = threadIdx.x & 31;
  if (wid >= NN) return;
  v4f a = *(const v4f*)(z  + (size_t)wid * 128 + lane * 4);
  v4f w = *(const v4f*)(Wd + lane * 4);
  float sum = a.x * w.x + a.y * w.y + a.z * w.z + a.w * w.w;
#pragma unroll
  for (int off = 16; off > 0; off >>= 1) sum += __shfl_xor(sum, off, 32);
  if (lane == 0) out[wid] = sum + bd[0];
}

extern "C" void kernel_launch(void* const* d_in, const int* in_sizes, int n_in,
                              void* d_out, int out_size, void* d_ws, size_t ws_size,
                              hipStream_t stream) {
  const float* x       = (const float*)d_in[0];
  const int*   ei      = (const int*)d_in[1];
  const float* eps     = (const float*)d_in[2];
  const float* enc_Wl  = (const float*)d_in[3];
  const float* enc_Wr  = (const float*)d_in[4];
  const float* enc_att = (const float*)d_in[5];
  const float* enc_b   = (const float*)d_in[6];
  const float* dec_Wl  = (const float*)d_in[7];
  const float* dec_Wr  = (const float*)d_in[8];
  const float* dec_att = (const float*)d_in[9];
  const float* dec_b   = (const float*)d_in[10];
  const float* W_mu    = (const float*)d_in[11];
  const float* b_mu    = (const float*)d_in[12];
  const float* W_ls    = (const float*)d_in[13];
  const float* b_ls    = (const float*)d_in[14];
  const float* W_out   = (const float*)d_in[15];
  const float* b_out   = (const float*)d_in[16];
  const float* W_deg   = (const float*)d_in[17];
  const float* b_deg   = (const float*)d_in[18];

  const int* src = ei;
  const int* dst = ei + EE;

  char* w = (char*)d_ws;
  const size_t nd = (size_t)NN * DD * sizeof(float);
  float*    h     = (float*)w;    w += nd;
  float*    xl    = (float*)w;    w += nd;
  float*    xr    = (float*)w;    w += nd;
  float*    hn    = (float*)w;    w += nd;
  float*    ewv   = (float*)w;    w += (size_t)ETOT * sizeof(float);
  unsigned* mord  = (unsigned*)w; w += (size_t)NN * sizeof(unsigned);
  float*    denom = (float*)w;    w += (size_t)NN * sizeof(float);

  float* xrec = (float*)d_out;
  float* zout = xrec + (size_t)NN * DD;
  float* muo  = zout + (size_t)NN * DD;
  float* lso  = muo  + (size_t)NN * DD;
  float* dego = lso  + (size_t)NN * DD;

  const dim3 b256(256);
  const int gGemm = (NN + ROWS - 1) / ROWS;
  const int gEW   = (ETOT * 32 + 255) / 256;   // one wave per edge
  const int gE    = (ETOT + 255) / 256;
  const int gND   = (NN * DD + 255) / 256;
  const int gNW   = (NN * 32 + 255) / 256;

  hipMemcpyAsync(h, x, nd, hipMemcpyDeviceToDevice, stream);

  for (int l = 0; l < LL; ++l) {
    gemm128<<<gGemm, b256, 0, stream>>>(h, enc_Wl + (size_t)l * DD * DD, nullptr, xl, NN);
    gemm128<<<gGemm, b256, 0, stream>>>(h, enc_Wr + (size_t)l * DD * DD, nullptr, xr, NN);
    layer_init<<<gND, b256, 0, stream>>>(enc_b + (size_t)l * DD, hn, mord, denom);
    edge_scores<<<gEW, b256, 0, stream>>>(xl, xr, enc_att + (size_t)l * DD, src, dst, ewv, mord);
    edge_norm<<<gE, b256, 0, stream>>>(dst, mord, ewv, denom);
    edge_agg<<<gEW, b256, 0, stream>>>(xl, src, dst, ewv, denom, hn);
    gelu_k<<<gND, b256, 0, stream>>>(hn, h, NN * DD);
  }

  gemm128<<<gGemm, b256, 0, stream>>>(h, W_mu, b_mu, muo, NN);
  gemm128<<<gGemm, b256, 0, stream>>>(h, W_ls, b_ls, lso, NN);
  vae_k<<<gND, b256, 0, stream>>>(muo, lso, eps, zout, NN * DD);

  hipMemcpyAsync(h, zout, nd, hipMemcpyDeviceToDevice, stream);
  for (int l = 0; l < LL; ++l) {
    gemm128<<<gGemm, b256, 0, stream>>>(h, dec_Wl + (size_t)l * DD * DD, nullptr, xl, NN);
    gemm128<<<gGemm, b256, 0, stream>>>(h, dec_Wr + (size_t)l * DD * DD, nullptr, xr, NN);
    layer_init<<<gND, b256, 0, stream>>>(dec_b + (size_t)l * DD, hn, mord, denom);
    edge_scores<<<gEW, b256, 0, stream>>>(xl, xr, dec_att + (size_t)l * DD, src, dst, ewv, mord);
    edge_norm<<<gE, b256, 0, stream>>>(dst, mord, ewv, denom);
    edge_agg<<<gEW, b256, 0, stream>>>(xl, src, dst, ewv, denom, hn);
    gelu_k<<<gND, b256, 0, stream>>>(hn, h, NN * DD);
  }

  gemm128<<<gGemm, b256, 0, stream>>>(h, W_out, b_out, xrec, NN);
  deg_k<<<gNW, b256, 0, stream>>>(zout, W_deg, b_deg, dego);

  (void)in_sizes; (void)n_in; (void)out_size; (void)ws_size;
}
